// UserModel_90125593740106
// MI455X (gfx1250) — matrix operand order repaired and checked by reading, not compiled, not run on hardware
//
#include <hip/hip_runtime.h>
#include <hip/hip_bf16.h>
#include <math.h>

// Problem constants (from reference)
constexpr int BB  = 128;   // batch
constexpr int SS  = 200;   // sequence
constexpr int VV  = 64;    // hidden
constexpr int NCC = 4000;  // C3 table width

typedef __attribute__((ext_vector_type(16))) _Float16 v16h;
typedef __attribute__((ext_vector_type(8)))  _Float16 v8h;
typedef __attribute__((ext_vector_type(8)))  float    v8f;
typedef __attribute__((ext_vector_type(4)))  unsigned int v4u;
typedef __attribute__((ext_vector_type(8)))  int      v8i;
typedef __attribute__((ext_vector_type(4)))  int      v4i;

// ---------------------------------------------------------------------------
// WMMA helpers (CDNA5 wave32, V_WMMA_F32_16X16X32_F16)
// ---------------------------------------------------------------------------
__device__ __forceinline__ v8f wmma16x16x32(v16h a, v16h b, v8f c) {
  // (neg_a, A, neg_b, B, c_mod, C, reuse_a, reuse_b)
  return __builtin_amdgcn_wmma_f32_16x16x32_f16(false, a, false, b, (short)0, c,
                                                false, false);
}

// A fragment: 16x32 f16, row-major in LDS with leading dim `ld` (halves).
// Lanes 0-15: M=lane, K 0..7 (v0-3) and 16..23 (v4-7)
// Lanes 16-31: M=lane-16, K 8..15 and 24..31
__device__ __forceinline__ v16h load_frag_a(const _Float16* p, int ld, int lane) {
  int m  = lane & 15;
  int k0 = (lane & 16) ? 8 : 0;
  const _Float16* r = p + m * ld + k0;
  v8h lo = *(const v8h*)(r);
  v8h hi = *(const v8h*)(r + 16);
  v16h a;
#pragma unroll
  for (int i = 0; i < 8; ++i) { a[i] = lo[i]; a[8 + i] = hi[i]; }
  return a;
}

// B fragment for out = X @ W^T : B[k][n] = W[nbase+n][k], W row-major [N][K]
// Lanes 0-15: col n=lane, K 0..7 and 16..23 ; lanes 16-31: col n=lane-16, K 8..15 and 24..31
__device__ __forceinline__ v16h load_frag_b(const _Float16* p, int ld, int lane) {
  int n  = lane & 15;
  int k0 = (lane & 16) ? 8 : 0;
  const _Float16* r = p + n * ld + k0;
  v8h lo = *(const v8h*)(r);
  v8h hi = *(const v8h*)(r + 16);
  v16h b;
#pragma unroll
  for (int i = 0; i < 8; ++i) { b[i] = lo[i]; b[8 + i] = hi[i]; }
  return b;
}
// C/D layout: element v of v8f at lane l -> row = v + 8*(l>>4), col = l&15

// ---------------------------------------------------------------------------
// TDM: DMA one step's xg slab (128*192 f16 = 49152 B) from global into LDS.
// D# built per CDNA5 ISA 8.3/8.4: 1-D tile, data_size=8B, 6144 units.
// Issued by one wave; completion via TENSORcnt.
// This toolchain declares the 6-arg builtin:
//   (uint32x4 g0, int32x8 g1, int32x4, int32x4, int32x8, i32 cpol)
// ---------------------------------------------------------------------------
__device__ __forceinline__ void tdm_load_xg_slab(const _Float16* src,
                                                 unsigned lds_byte_off) {
  unsigned long long ga = (unsigned long long)(uintptr_t)src;
  v4u g0;
  g0[0] = 1u;                                  // count=1, load, user mode
  g0[1] = lds_byte_off;                        // lds_addr
  g0[2] = (unsigned)ga;                        // global_addr[31:0]
  g0[3] = ((unsigned)(ga >> 32) & 0x01FFFFFFu) // global_addr[56:32]
          | (2u << 30);                        // type = 2 ("image")
  v8i g1;
  g1[0] = (int)(3u << 16);                     // data_size = 3 -> 8 bytes
  g1[1] = (int)((6144u & 0xFFFFu) << 16);      // tensor_dim0[15:0] @ bits63:48
  g1[2] = (int)((6144u >> 16) | (1u << 16));   // tensor_dim0[31:16], tensor_dim1=1
  g1[3] = (int)(6144u << 16);                  // tile_dim0 = 6144 @ bits127:112
  g1[4] = 0;                                   // tile_dim1=0, tile_dim2=0
  g1[5] = (int)6144u;                          // tensor_dim0_stride lo32
  g1[6] = 0;
  g1[7] = 0;
  v4i z4 = {0, 0, 0, 0};
  v8i z8 = {0, 0, 0, 0, 0, 0, 0, 0};
  __builtin_amdgcn_tensor_load_to_lds(g0, g1, z4, z4, z8, 0);
}

// ---------------------------------------------------------------------------
// Kernel 1: xg = [gamma*v_d | R_w[r]] @ W_ih^T + b_ih   for all (b,s)
//   M = B*S = 25600 (16-row tiles), K = 128, N = 192
//   Output layout: f16, [s][b][192] so each step's slab is contiguous (TDM src)
// ---------------------------------------------------------------------------
__global__ __launch_bounds__(128) void embed_xg_kernel(
    const int* __restrict__ d_seq, const int* __restrict__ r_seq,
    const float* __restrict__ D_w, const float* __restrict__ v_d,
    const float* __restrict__ R_w, const float* __restrict__ W_ih,
    const float* __restrict__ b_ih, _Float16* __restrict__ xg16) {
  __shared__ _Float16 sWih[192 * 136];
  __shared__ _Float16 sX[16 * 136];
  __shared__ float    sBih[192];

  int tid = threadIdx.x;
  for (int e = tid; e < 192 * 128; e += 128)
    sWih[(e >> 7) * 136 + (e & 127)] = (_Float16)W_ih[e];
  for (int e = tid; e < 192; e += 128) sBih[e] = b_ih[e];
  __syncthreads();

  int wave = tid >> 5, lane = tid & 31;
  int n = lane & 15, hh = lane >> 4;

  for (int tile = blockIdx.x; tile < (BB * SS) / 16; tile += gridDim.x) {
    // stage 16 rows of x = [gamma*v_d | R_w[r]] (f32 -> f16)
    for (int e = tid; e < 16 * 128; e += 128) {
      int rr = e >> 7, cc = e & 127;
      int g = tile * 16 + rr;  // g = b*S + s
      float val;
      if (cc < 64) val = D_w[d_seq[g]] * v_d[cc];
      else         val = R_w[r_seq[g] * 64 + (cc - 64)];
      sX[rr * 136 + cc] = (_Float16)val;
    }
    __syncthreads();

    v8f acc0 = {}, acc1 = {}, acc2 = {};
#pragma unroll
    for (int ks = 0; ks < 4; ++ks) {
      v16h a  = load_frag_a(sX + ks * 32, 136, lane);
      v16h b0 = load_frag_b(sWih + ((wave * 3 + 0) * 16) * 136 + ks * 32, 136, lane);
      v16h b1 = load_frag_b(sWih + ((wave * 3 + 1) * 16) * 136 + ks * 32, 136, lane);
      v16h b2 = load_frag_b(sWih + ((wave * 3 + 2) * 16) * 136 + ks * 32, 136, lane);
      acc0 = wmma16x16x32(a, b0, acc0);
      acc1 = wmma16x16x32(a, b1, acc1);
      acc2 = wmma16x16x32(a, b2, acc2);
    }
#pragma unroll
    for (int v = 0; v < 8; ++v) {
      int g = tile * 16 + v + 8 * hh;  // flattened b*S + s
      int bidx = g / SS, s = g % SS;
      size_t rowoff = ((size_t)s * BB + bidx) * 192;
      int c0 = (wave * 3 + 0) * 16 + n;
      int c1 = (wave * 3 + 1) * 16 + n;
      int c2 = (wave * 3 + 2) * 16 + n;
      xg16[rowoff + c0] = (_Float16)(acc0[v] + sBih[c0]);
      xg16[rowoff + c1] = (_Float16)(acc1[v] + sBih[c1]);
      xg16[rowoff + c2] = (_Float16)(acc2[v] + sBih[c2]);
    }
    __syncthreads();
  }
}

// ---------------------------------------------------------------------------
// Kernel 2: sequential scan, single persistent workgroup (32 waves)
// ---------------------------------------------------------------------------
constexpr int WHH_LD = 72;   // W_hh  [192][64] f16
constexpr int W1_LD  = 72;   // l1_w1 [64][64]  f16
constexpr int W2_LD  = 200;  // l2_w1 [64][192] f16
constexpr int H_LD   = 72;   // h     [128][64] f16
constexpr int Z_LD   = 192;  // z/xg  [128][192] f16 (contiguous: TDM target)

constexpr int SZ_OFF_H = 192 * WHH_LD + 64 * W1_LD + 64 * W2_LD + 128 * H_LD;
constexpr unsigned SZ_OFF_BYTES = (unsigned)SZ_OFF_H * 2;

constexpr int SCAN_H16 = SZ_OFF_H + 128 * Z_LD;
constexpr int SCAN_F32 = 192 + 64 + 64 + 64 + 64 + 64 + 64 + 128 + 128 + 128 + 128 + 128 + 128 + 4;
constexpr int SCAN_I32 = 256;
constexpr size_t SCAN_LDS_BYTES = (size_t)SCAN_H16 * 2 + (size_t)(SCAN_F32 + SCAN_I32) * 4;

__global__ __launch_bounds__(1024) void scan_kernel(
    const int* __restrict__ c3_seq, const int* __restrict__ d_seq,
    const int* __restrict__ r_seq, const float* __restrict__ v_c3,
    const float* __restrict__ D_w, const float* __restrict__ v_d,
    const float* __restrict__ R_w, const float* __restrict__ W_hh,
    const float* __restrict__ b_hh, const float* __restrict__ l1_w1,
    const float* __restrict__ l1_b1, const float* __restrict__ l1_w2,
    const float* __restrict__ l1_b2, const float* __restrict__ l2_w1,
    const float* __restrict__ l2_b1, const float* __restrict__ l2_w2,
    const float* __restrict__ l2_b2, const _Float16* __restrict__ xg16,
    float* __restrict__ C3ws, float* __restrict__ newc3v,
    float* __restrict__ out_alpha, float* __restrict__ out_h) {
  extern __shared__ char smem[];
  _Float16* sWhh = (_Float16*)smem;            // 192 x 72
  _Float16* sW1  = sWhh + 192 * WHH_LD;        // 64 x 72
  _Float16* sW2  = sW1 + 64 * W1_LD;           // 64 x 200
  _Float16* sH   = sW2 + 64 * W2_LD;           // 128 x 72
  _Float16* sZ   = sH + 128 * H_LD;            // 128 x 192 : xg(t) then z(t)
  float* fb        = (float*)(sZ + 128 * Z_LD);
  float* sBhh      = fb;               // 192
  float* sB1       = sBhh + 192;       // 64
  float* sl1w2     = sB1 + 64;         // 64
  float* sB2       = sl1w2 + 64;       // 64
  float* sl2w2     = sB2 + 64;         // 64
  float* svc3      = sl2w2 + 64;       // 64
  float* svd       = svc3 + 64;        // 64
  float* sRw       = svd + 64;         // 128
  float* sAlpha    = sRw + 128;        // 128
  float* sAlphaAcc = sAlpha + 128;     // 128
  float* sC3Acc    = sAlphaAcc + 128;  // 128
  float* sBeta     = sC3Acc + 128;     // 128
  float* sGam      = sBeta + 128;      // 128
  float* sConst    = sGam + 128;       // 4 (2 used)
  int* sC3i  = (int*)(sConst + 4);     // 128
  int* sRsel = sC3i + 128;             // 128

  int tid = threadIdx.x;
  int wave = tid >> 5, lane = tid & 31;

  // --- init: weights f32->f16 into LDS, zero state ---
  for (int e = tid; e < 192 * 64; e += 1024)
    sWhh[(e >> 6) * WHH_LD + (e & 63)] = (_Float16)W_hh[e];
  for (int e = tid; e < 64 * 64; e += 1024)
    sW1[(e >> 6) * W1_LD + (e & 63)] = (_Float16)l1_w1[e];
  for (int e = tid; e < 64 * 192; e += 1024)
    sW2[(e / 192) * W2_LD + (e % 192)] = (_Float16)l2_w1[e];
  for (int e = tid; e < 128 * H_LD; e += 1024) sH[e] = (_Float16)0.0f;
  if (tid < 192) sBhh[tid] = b_hh[tid];
  if (tid < 64) {
    sB1[tid] = l1_b1[tid]; sl1w2[tid] = l1_w2[tid];
    sB2[tid] = l2_b1[tid]; sl2w2[tid] = l2_w2[tid];
    svc3[tid] = v_c3[tid]; svd[tid] = v_d[tid];
  }
  if (tid < 128) { sRw[tid] = R_w[tid]; sAlpha[tid] = 0.0f; }
  if (tid == 0) { sConst[0] = l1_b2[0]; sConst[1] = l2_b2[0]; }
  for (int e = tid; e < BB * NCC; e += 1024) C3ws[e] = 0.0f;  // C3 state = 0

  // dynamic LDS starts after static LDS (0 here); TDM lds_addr is absolute
  unsigned sz_lds = SZ_OFF_BYTES + (unsigned)__builtin_amdgcn_groupstaticsize();

  // preload xg(t=0) slab via Tensor Data Mover (wave 0 only, one DMA op)
  if (wave == 0) {
    tdm_load_xg_slab(xg16, sz_lds);
    __builtin_amdgcn_s_wait_tensorcnt(0);
  }
  __syncthreads();

  int i = wave >> 2;        // row tile 0..7  (rows i*16..i*16+15)
  int j = wave & 3;         // col tile 0..3  (cols of V=64)
  int n = lane & 15, hh = lane >> 4;
  int col = j * 16 + n;
  int rowbase = i * 16 + 8 * hh;

  v8f hreg = {};  // persistent fp32 h tile (this wave's rows/cols)

  for (int t = 0; t < SS; ++t) {
    // ---- phase A: GRU gates via WMMA  (hg = h@W_hh^T) ----
    v8f accR = {}, accZ = {}, accN = {};
#pragma unroll
    for (int ks = 0; ks < 2; ++ks) {
      v16h a  = load_frag_a(sH + (i * 16) * H_LD + ks * 32, H_LD, lane);
      v16h bR = load_frag_b(sWhh + (0   + j * 16) * WHH_LD + ks * 32, WHH_LD, lane);
      v16h bZ = load_frag_b(sWhh + (64  + j * 16) * WHH_LD + ks * 32, WHH_LD, lane);
      v16h bN = load_frag_b(sWhh + (128 + j * 16) * WHH_LD + ks * 32, WHH_LD, lane);
      accR = wmma16x16x32(a, bR, accR);
      accZ = wmma16x16x32(a, bZ, accZ);
      accN = wmma16x16x32(a, bN, accN);
    }
    // TDM slab for this t was issued last step (or preloop); make it visible
    if (wave == 0) __builtin_amdgcn_s_wait_tensorcnt(0);
    __syncthreads();
#pragma unroll
    for (int v = 0; v < 8; ++v) {
      int b = rowbase + v;
      const _Float16* xrow = sZ + b * Z_LD;  // xg(t) staged in LDS
      float hr = accR[v] + sBhh[col];
      float hz = accZ[v] + sBhh[64 + col];
      float hn = accN[v] + sBhh[128 + col];
      float rg = 1.0f / (1.0f + expf(-((float)xrow[col] + hr)));
      float zg = 1.0f / (1.0f + expf(-((float)xrow[64 + col] + hz)));
      float ng = tanhf((float)xrow[128 + col] + rg * hn);
      hreg[v] = (1.0f - zg) * ng + zg * hreg[v];
    }
    __syncthreads();  // all WMMA reads of old sH complete

    // ---- phase B: publish h (f16 to LDS, f32 to global); fetch scalars ----
#pragma unroll
    for (int v = 0; v < 8; ++v) {
      int b = rowbase + v;
      sH[b * H_LD + col] = (_Float16)hreg[v];
      out_h[((size_t)b * SS + t) * VV + col] = hreg[v];
    }
    if (tid < 128) {
      int b = tid;
      int c3i = c3_seq[b * SS + t];
      sC3i[b]  = c3i;
      sRsel[b] = r_seq[b * SS + t];
      sGam[b]  = D_w[d_seq[b * SS + t]];
      sBeta[b] = C3ws[b * NCC + c3i];  // gather BEFORE this step's scatter
      sAlphaAcc[b] = 0.0f;
      sC3Acc[b]    = 0.0f;
    }
    __syncthreads();

    // ---- phase C: mlp1 = relu(h@l1_w1^T + b1) . l1_w2 ----
    {
      v8f acc = {};
#pragma unroll
      for (int ks = 0; ks < 2; ++ks) {
        v16h a  = load_frag_a(sH + (i * 16) * H_LD + ks * 32, H_LD, lane);
        v16h bw = load_frag_b(sW1 + (j * 16) * W1_LD + ks * 32, W1_LD, lane);
        acc = wmma16x16x32(a, bw, acc);
      }
#pragma unroll
      for (int v = 0; v < 8; ++v) {
        float val = fmaxf(acc[v] + sB1[col], 0.0f) * sl1w2[col];
#pragma unroll
        for (int msk = 1; msk < 16; msk <<= 1) val += __shfl_xor(val, msk, 32);
        if (n == 0) atomicAdd(&sAlphaAcc[rowbase + v], val);
      }
    }
    __syncthreads();

    // ---- phase D: alpha rule + build z = [beta*v_c3 | gamma*v_d | R_w[r]] ----
    if (tid < 128) {
      int b = tid;
      float an  = sAlphaAcc[b] + sConst[0];
      float old = sAlpha[b];
      bool cond = (old - sGam[b]) >= 0.0f;
      float na  = (sRsel[b] == 1) ? (cond ? an : old) : (cond ? old : an);
      sAlpha[b] = na;
      out_alpha[(size_t)b * SS + t] = na;
    }
    for (int e = tid; e < 128 * 192; e += 1024) {
      int b = e / 192, c = e % 192;
      float val;
      if (c < 64)       val = sBeta[b] * svc3[c];
      else if (c < 128) val = sGam[b] * svd[c - 64];
      else              val = sRw[sRsel[b] * 64 + (c - 128)];
      sZ[b * Z_LD + c] = (_Float16)val;  // overwrite consumed xg(t)
    }
    __syncthreads();

    // ---- phase E: mlp2 = relu(z@l2_w1^T + b1) . l2_w2 ----
    {
      v8f acc = {};
#pragma unroll
      for (int ks = 0; ks < 6; ++ks) {
        v16h a  = load_frag_a(sZ + (i * 16) * Z_LD + ks * 32, Z_LD, lane);
        v16h bw = load_frag_b(sW2 + (j * 16) * W2_LD + ks * 32, W2_LD, lane);
        acc = wmma16x16x32(a, bw, acc);
      }
#pragma unroll
      for (int v = 0; v < 8; ++v) {
        float val = fmaxf(acc[v] + sB2[col], 0.0f) * sl2w2[col];
#pragma unroll
        for (int msk = 1; msk < 16; msk <<= 1) val += __shfl_xor(val, msk, 32);
        if (n == 0) atomicAdd(&sC3Acc[rowbase + v], val);
      }
    }
    __syncthreads();

    // ---- phase F: scatter into C3 state; kick TDM for xg(t+1) ----
    if (tid < 128) {
      int b = tid;
      float nv = sC3Acc[b] + sConst[1];
      C3ws[b * NCC + sC3i[b]] = nv;
      newc3v[t * BB + b] = nv;
    }
    if (wave == 0 && t + 1 < SS)
      tdm_load_xg_slab(xg16 + (size_t)(t + 1) * BB * 192, sz_lds);
    __syncthreads();
  }
}

// ---------------------------------------------------------------------------
// Kernel 3: reconstruct C3_seq[b,t,j] from scatter history (bandwidth bound)
// ---------------------------------------------------------------------------
__global__ __launch_bounds__(256) void reconstruct_c3_kernel(
    const int* __restrict__ c3_seq, const float* __restrict__ newc3v,
    float* __restrict__ out_c3) {
  __shared__ int   sIdx[SS];
  __shared__ float sVal[SS];
  int b  = blockIdx.x >> 4;   // 128 batch rows
  int jc = blockIdx.x & 15;   // 16 chunks of 256 columns
  int tid = threadIdx.x;
  if (tid < SS) {
    sIdx[tid] = c3_seq[b * SS + tid];
    sVal[tid] = newc3v[tid * BB + b];
  }
  __syncthreads();
  int j = jc * 256 + tid;
  if (j >= NCC) return;
  float cur = 0.0f;
  float* dst = out_c3 + (size_t)b * SS * NCC + j;
#pragma unroll 4
  for (int t = 0; t < SS; ++t) {
    if (sIdx[t] == j) cur = sVal[t];
    dst[(size_t)t * NCC] = cur;  // coalesced across j
  }
}

// ---------------------------------------------------------------------------
// Host entry
// ---------------------------------------------------------------------------
extern "C" void kernel_launch(void* const* d_in, const int* in_sizes, int n_in,
                              void* d_out, int out_size, void* d_ws,
                              size_t ws_size, hipStream_t stream) {
  (void)in_sizes; (void)n_in; (void)out_size;
  const int*   c3_seq = (const int*)d_in[0];
  const int*   d_seq  = (const int*)d_in[1];
  const int*   r_seq  = (const int*)d_in[2];
  const float* v_c3   = (const float*)d_in[3];
  const float* D_w    = (const float*)d_in[4];
  const float* v_d    = (const float*)d_in[5];
  const float* R_w    = (const float*)d_in[6];
  const float* W_ih   = (const float*)d_in[7];
  const float* W_hh   = (const float*)d_in[8];
  const float* b_ih   = (const float*)d_in[9];
  const float* b_hh   = (const float*)d_in[10];
  const float* l1_w1  = (const float*)d_in[11];
  const float* l1_b1  = (const float*)d_in[12];
  const float* l1_w2  = (const float*)d_in[13];
  const float* l1_b2  = (const float*)d_in[14];
  const float* l2_w1  = (const float*)d_in[15];
  const float* l2_b1  = (const float*)d_in[16];
  const float* l2_w2  = (const float*)d_in[17];
  const float* l2_b2  = (const float*)d_in[18];

  float* out       = (float*)d_out;
  float* out_alpha = out;                                   // [B,S]
  float* out_h     = out + (size_t)BB * SS;                 // [B,S,V]
  float* out_c3    = out_h + (size_t)BB * SS * VV;          // [B,S,NC3]

  // workspace layout: C3 state (f32) | new_c3 (f32) | xg slabs (f16, [s][b][192])
  const size_t c3ws_n = (size_t)BB * NCC;        //   512,000 f32
  const size_t nc3_n  = (size_t)SS * BB;         //    25,600 f32
  const size_t xg_n   = (size_t)SS * BB * 192;   // 4,915,200 f16
  const size_t need = (c3ws_n + nc3_n) * sizeof(float) + xg_n * sizeof(_Float16);
  if (ws_size < need) return;  // ~12.1 MB
  float*    C3ws   = (float*)d_ws;
  float*    newc3v = C3ws + c3ws_n;
  _Float16* xg16   = (_Float16*)(newc3v + nc3_n);

  // 1) parallel WMMA precompute of GRU input gates (f16 slabs per step)
  embed_xg_kernel<<<200, 128, 0, stream>>>(d_seq, r_seq, D_w, v_d, R_w, W_ih,
                                           b_ih, xg16);
  // 2) sequential scan: one persistent 32-wave workgroup, ~136 KB LDS,
  //    per-step xg slab delivered by the Tensor Data Mover
  scan_kernel<<<1, 1024, SCAN_LDS_BYTES, stream>>>(
      c3_seq, d_seq, r_seq, v_c3, D_w, v_d, R_w, W_hh, b_hh, l1_w1, l1_b1,
      l1_w2, l1_b2, l2_w1, l2_b1, l2_w2, l2_b2, xg16, C3ws, newc3v, out_alpha,
      out_h);
  // 3) bandwidth-bound C3_seq snapshot reconstruction (409.6 MB of stores)
  reconstruct_c3_kernel<<<BB * 16, 256, 0, stream>>>(c3_seq, newc3v, out_c3);
}
